// SelCondPredictor_46763603918972
// MI455X (gfx1250) — compile-verified, matrix operand order checked by reading, not run on hardware
//
#include <hip/hip_runtime.h>
#include <math.h>

// ---------------------------------------------------------------------------
// SelCondPredictor forward for MI455X (gfx1250), wave32 + WMMA f16->f32.
// All matmuls (LSTM input projections, recurrent h@Wh.T, attention einsums,
// output MLP projections) run through v_wmma_f32_16x16x32_f16.
// Staging is fully vectorized (b128 global loads / b128 LDS ops), and uses
// GLOBAL_LOAD_ASYNC_TO_LDS_B128 when the toolchain exposes the builtin.
// Requires K % 32 == 0 and 16B-aligned rows, which all call sites satisfy.
// ---------------------------------------------------------------------------

typedef __attribute__((ext_vector_type(16))) _Float16 v16h;
typedef __attribute__((ext_vector_type(8)))  _Float16 v8h;
typedef __attribute__((ext_vector_type(8)))  float    v8f;
typedef __attribute__((ext_vector_type(4)))  int      v4i;

#if defined(__has_builtin)
#if __has_builtin(__builtin_amdgcn_global_load_async_to_lds_b128) && \
    __has_builtin(__builtin_amdgcn_s_wait_asynccnt)
#define HAVE_ASYNC_LDS 1
#endif
#endif
#ifndef HAVE_ASYNC_LDS
#define HAVE_ASYNC_LDS 0
#endif

#if HAVE_ASYNC_LDS
__device__ __forceinline__ void async_ld128(const _Float16* g, _Float16* l) {
  // Toolchain signature (from diagnostics): generic int4* parameters.
  __builtin_amdgcn_global_load_async_to_lds_b128((v4i*)g, (v4i*)l, 0, 0);
}
#endif

#define NEGV (-100.0f)
#define NB   128   // batch
#define NTX  128   // question tokens
#define NTC  32    // columns
#define ND   512   // N_h
#define NHID 256   // per-direction hidden
#define NG   1024  // 4*H gates

__device__ __forceinline__ float sigf(float x) { return 1.0f / (1.0f + __expf(-x)); }

__device__ __forceinline__ v16h pack16(v8h lo, v8h hi) {
  return __builtin_shufflevector(lo, hi, 0, 1, 2, 3, 4, 5, 6, 7,
                                         8, 9, 10, 11, 12, 13, 14, 15);
}

// ---------------------------------------------------------------------------
// Elementwise converts / concat
// ---------------------------------------------------------------------------
__global__ void k_f32_to_f16(const float* __restrict__ src, _Float16* __restrict__ dst, int n) {
  int i = blockIdx.x * blockDim.x + threadIdx.x;
  if (i < n) dst[i] = (_Float16)src[i];
}

__global__ void k_concat_h(const float* __restrict__ a, const float* __restrict__ b,
                           _Float16* __restrict__ dst, int rows, int d) {
  int i = blockIdx.x * blockDim.x + threadIdx.x;
  int n = rows * 2 * d;
  if (i >= n) return;
  int r = i / (2 * d), c = i % (2 * d);
  float v = (c < d) ? a[(long)r * d + c] : b[(long)r * d + (c - d)];
  dst[i] = (_Float16)v;
}

// ---------------------------------------------------------------------------
// Generic batched WMMA GEMM:  C[M,N] = A[M,K](f16) * op(B)(f16) + bias
//   b_kxn==0 : B stored [N,K] row-major (weight-like, computes A @ B^T)
//   b_kxn==1 : B stored [K,N] row-major (plain A @ B)
// Block tile 64x64x32, 4 waves (2x2), each wave 32x32 (2x2 WMMA tiles).
// LDS rows padded to 40 halves (80B = 5x16B) so all fragment reads are
// aligned ds_load_b128. Full tiles stage via async global->LDS copies.
// ---------------------------------------------------------------------------
#define BM 64
#define BN 64
#define BK 32
#define LP 40   // padded LDS row length in halves

__global__ __launch_bounds__(128) void k_gemm(
    const _Float16* __restrict__ A, const _Float16* __restrict__ Bm,
    const float* __restrict__ bias, float* __restrict__ C,
    int M, int N, int K, int lda, int ldb, int ldc,
    long sA, long sB, long sC, int b_kxn)
{
  __shared__ _Float16 As[BM][LP];
  __shared__ _Float16 Bs[BN][LP];
  int z = blockIdx.z;
  A  += (long)z * sA;
  Bm += (long)z * sB;
  C  += (long)z * sC;
  int m0 = blockIdx.y * BM, n0 = blockIdx.x * BN;
  int tid = threadIdx.x, lane = tid & 31, wv = tid >> 5;
  int hf = lane >> 4, l15 = lane & 15;
  int wm = (wv >> 1) * 32, wn = (wv & 1) * 32;
  int kb0 = hf * 8, kb1 = 16 + hf * 8;
  const v8h zv = {};
  const bool fullM = (m0 + BM <= M);
  const bool fullN = (n0 + BN <= N);

  v8f acc[2][2] = {};

  for (int k0 = 0; k0 < K; k0 += BK) {
    // ---- stage A tile: 64x32 halves = 256 chunks of 8; 2 iters x 128 thr
#if HAVE_ASYNC_LDS
    if (fullM) {
      #pragma unroll
      for (int i = 0; i < 2; ++i) {
        int cidx = tid + i * 128;
        int r = cidx >> 2, kc = (cidx & 3) * 8;
        async_ld128(&A[(long)(m0 + r) * lda + k0 + kc], &As[r][kc]);
      }
    } else
#endif
    {
      #pragma unroll
      for (int i = 0; i < 2; ++i) {
        int cidx = tid + i * 128;
        int r = cidx >> 2, kc = (cidx & 3) * 8;
        v8h v = zv;
        if (m0 + r < M) v = *(const v8h*)&A[(long)(m0 + r) * lda + k0 + kc];
        *(v8h*)&As[r][kc] = v;
      }
    }
    // ---- stage B tile ----
    if (b_kxn) {
      // B[K,N] row-major: vector load along N, transpose into Bs[n][k]
      #pragma unroll
      for (int i = 0; i < 2; ++i) {
        int cidx = tid + i * 128;
        int kk = cidx >> 3, nc = (cidx & 7) * 8;
        v8h v = *(const v8h*)&Bm[(long)(k0 + kk) * ldb + n0 + nc];
        #pragma unroll
        for (int q = 0; q < 8; ++q) Bs[nc + q][kk] = v[q];
      }
    } else {
#if HAVE_ASYNC_LDS
      if (fullN) {
        #pragma unroll
        for (int i = 0; i < 2; ++i) {
          int cidx = tid + i * 128;
          int nn = cidx >> 2, kc = (cidx & 3) * 8;
          async_ld128(&Bm[(long)(n0 + nn) * ldb + kc + k0], &Bs[nn][kc]);
        }
      } else
#endif
      {
        // B[N,K] row-major (weight): contiguous along K
        #pragma unroll
        for (int i = 0; i < 2; ++i) {
          int cidx = tid + i * 128;
          int nn = cidx >> 2, kc = (cidx & 3) * 8;
          v8h v = zv;
          if (n0 + nn < N) v = *(const v8h*)&Bm[(long)(n0 + nn) * ldb + kc + k0];
          *(v8h*)&Bs[nn][kc] = v;
        }
      }
    }
    // prefetch next K tile into cache (global_prefetch_b8)
    if (k0 + BK < K) {
      if (m0 + (tid >> 1) < M)
        __builtin_prefetch(&A[(long)(m0 + (tid >> 1)) * lda + k0 + BK], 0, 1);
    }
#if HAVE_ASYNC_LDS
    __builtin_amdgcn_s_wait_asynccnt(0);   // LDS writes from async copies done
#endif
    __syncthreads();

    // ---- fragments per CDNA5 16-bit A/B layouts (aligned b128 LDS reads)
    v16h afr[2], bfr[2];
    #pragma unroll
    for (int i = 0; i < 2; ++i) {
      const _Float16* ap = &As[wm + i * 16 + l15][0];
      afr[i] = pack16(*(const v8h*)&ap[kb0], *(const v8h*)&ap[kb1]);
      const _Float16* bp = &Bs[wn + i * 16 + l15][0];
      bfr[i] = pack16(*(const v8h*)&bp[kb0], *(const v8h*)&bp[kb1]);
    }
    #pragma unroll
    for (int i = 0; i < 2; ++i)
      #pragma unroll
      for (int j = 0; j < 2; ++j)
        acc[i][j] = __builtin_amdgcn_wmma_f32_16x16x32_f16(
            false, afr[i], false, bfr[j], (short)0, acc[i][j], false, false);
    __syncthreads();
  }

  // ---- store C (VGPR r -> M=r / r+8 per half, N = lane&15) ----
  #pragma unroll
  for (int i = 0; i < 2; ++i)
    #pragma unroll
    for (int j = 0; j < 2; ++j) {
      int mg = m0 + wm + i * 16 + hf * 8;
      int ng = n0 + wn + j * 16 + l15;
      #pragma unroll
      for (int r = 0; r < 8; ++r) {
        int row = mg + r;
        if (row < M && ng < N) {
          float v = acc[i][j][r];
          if (bias) v += bias[ng];
          C[(long)row * ldc + ng] = v;
        }
      }
    }
}

// ---------------------------------------------------------------------------
// LSTM scan: one workgroup = 16 batch rows, 256 threads = 8 waves.
// Per time step: g = gates_x[b,t,:] + h @ Wh^T  via WMMA (M=16,N=1024,K=256),
// then gate math (torch order i,f,g,o), h state in LDS (f16).
// Output written at reversed position for backward dir; padded slots zeroed
// (this *is* the reference's post-layer mask).
// ---------------------------------------------------------------------------
__global__ __launch_bounds__(256) void k_lstm_scan(
    const float* __restrict__ gates,   // [B, T, 1024] = x@Wi^T + b
    const _Float16* __restrict__ Wh,   // [1024, 256] row-major f16
    const int* __restrict__ lens,
    float* __restrict__ outp,          // [B, T, 512]
    int T, int colOff, int reverse)
{
  __shared__ _Float16 h_s[16][NHID + 8];   // row stride 264h = 528B (16B mult)
  __shared__ _Float16 g_s[16][NG + 8];     // row stride 1032h = 2064B
  int tid = threadIdx.x, lane = tid & 31, wv = tid >> 5;
  int hf = lane >> 4, l15 = lane & 15;
  int kb0 = hf * 8, kb1 = 16 + hf * 8;
  int b0 = blockIdx.x * 16;

  for (int i = tid; i < 16 * NHID; i += 256) h_s[i >> 8][i & 255] = (_Float16)0.f;
  float c_reg[16];
  #pragma unroll
  for (int j = 0; j < 16; ++j) c_reg[j] = 0.f;

  int row_gm = tid >> 4;          // gate-math row (0..15)
  int hbase  = (tid & 15) * 16;   // 16 contiguous hidden units per thread
  int len_gm = lens[b0 + row_gm];
  __syncthreads();

  for (int s = 0; s < T; ++s) {
    // ---- WMMA GEMM: g_rec = h(16x256) @ Wh^T(256x1024); wave owns 128 cols
    v8f acc[8] = {};
    #pragma unroll
    for (int ks = 0; ks < 8; ++ks) {
      const _Float16* ap = &h_s[l15][ks * 32];
      v16h afr = pack16(*(const v8h*)&ap[kb0], *(const v8h*)&ap[kb1]);
      #pragma unroll
      for (int nt = 0; nt < 8; ++nt) {
        int n = wv * 128 + nt * 16 + l15;   // gate column = row of Wh
        const _Float16* bp = Wh + (long)n * NHID + ks * 32;
        v16h bfr = pack16(*(const v8h*)&bp[kb0], *(const v8h*)&bp[kb1]);
        acc[nt] = __builtin_amdgcn_wmma_f32_16x16x32_f16(
            false, afr, false, bfr, (short)0, acc[nt], false, false);
      }
    }
    // spill recurrent gates to LDS for cross-wave gate math
    #pragma unroll
    for (int nt = 0; nt < 8; ++nt) {
      int col = wv * 128 + nt * 16 + l15;
      #pragma unroll
      for (int r = 0; r < 8; ++r)
        g_s[r + hf * 8][col] = (_Float16)acc[nt][r];
    }
    __syncthreads();

    // ---- gate math: c' = sig(f)*c + sig(i)*tanh(g); h = sig(o)*tanh(c')
    int tsrc = reverse ? ((s < len_gm) ? (len_gm - 1 - s) : s) : s;
    const float* gx = gates + ((long)(b0 + row_gm) * T + tsrc) * NG;
    float* op = outp + ((long)(b0 + row_gm) * T + tsrc) * ND + colOff;
    #pragma unroll
    for (int j = 0; j < 16; ++j) {
      int h_idx = hbase + j;
      float gi = gx[h_idx]           + (float)g_s[row_gm][h_idx];
      float gf = gx[NHID + h_idx]    + (float)g_s[row_gm][NHID + h_idx];
      float gg = gx[2*NHID + h_idx]  + (float)g_s[row_gm][2*NHID + h_idx];
      float go = gx[3*NHID + h_idx]  + (float)g_s[row_gm][3*NHID + h_idx];
      float c = sigf(gf) * c_reg[j] + sigf(gi) * tanhf(gg);
      c_reg[j] = c;
      float h = sigf(go) * tanhf(c);
      h_s[row_gm][h_idx] = (_Float16)h;     // raw state carries on
      op[h_idx] = (s < len_gm) ? h : 0.f;   // masked output
    }
    __syncthreads();
  }
}

// ---------------------------------------------------------------------------
// Masked softmax over last dim (L==128). x: [B, R, L]; t-mask by tlen[b];
// optional row mask (fill NEG for r>=rlen[b]) matching reference semantics.
// ---------------------------------------------------------------------------
__global__ __launch_bounds__(128) void k_softmax128(
    float* __restrict__ x, const int* __restrict__ tlen,
    const int* __restrict__ rlen, int R)
{
  __shared__ float red[128];
  int rowid = blockIdx.x;
  int b = rowid / R, r = rowid % R;
  float* p = x + (long)rowid * 128;
  int t = threadIdx.x;
  float v = p[t];
  if (t >= tlen[b]) v = NEGV;
  if (rlen && r >= rlen[b]) v = NEGV;
  red[t] = v; __syncthreads();
  for (int s = 64; s > 0; s >>= 1) { if (t < s) red[t] = fmaxf(red[t], red[t + s]); __syncthreads(); }
  float m = red[0]; __syncthreads();
  float e = __expf(v - m);
  red[t] = e; __syncthreads();
  for (int s = 64; s > 0; s >>= 1) { if (t < s) red[t] += red[t + s]; __syncthreads(); }
  p[t] = e / red[0];
}

// ---------------------------------------------------------------------------
// Small VALU helpers
// ---------------------------------------------------------------------------
__global__ void k_chosen(const float* __restrict__ e_col, const int* __restrict__ gt,
                         float* __restrict__ outp) {
  int b = blockIdx.x, d = threadIdx.x;
  outp[(long)b * ND + d] = e_col[((long)b * NTC + gt[b]) * ND + d];
}

// out[b, r] = dot(Mx[b*rowsPerB + r, :K], vx[b, :K])
__global__ void k_dotvec(const float* __restrict__ Mx, const float* __restrict__ vx,
                         float* __restrict__ outp, int rowsPerB, int K) {
  int i = blockIdx.x * blockDim.x + threadIdx.x;
  if (i >= NB * rowsPerB) return;
  int b = i / rowsPerB;
  const float* mp = Mx + (long)i * K;
  const float* vp = vx + (long)b * K;
  float s = 0.f;
  for (int d = 0; d < K; ++d) s += mp[d] * vp[d];
  outp[i] = s;
}

// s[b,t] = sum_c att[b,c,t]   (att: [B,32,128])
__global__ void k_colsum(const float* __restrict__ att, float* __restrict__ s) {
  int i = blockIdx.x * blockDim.x + threadIdx.x;
  if (i >= NB * 128) return;
  int b = i / 128, t = i % 128;
  float a = 0.f;
  for (int c = 0; c < NTC; ++c) a += att[((long)b * NTC + c) * 128 + t];
  s[i] = a;
}

// out[b,d] = sum_t w[b,t] * Hm[b,t,d]   (T==128, D==512)
__global__ void k_vecmat(const float* __restrict__ w, const float* __restrict__ Hm,
                         float* __restrict__ outp) {
  int i = blockIdx.x * blockDim.x + threadIdx.x;
  if (i >= NB * ND) return;
  int b = i >> 9, d = i & (ND - 1);
  float s = 0.f;
  for (int t = 0; t < 128; ++t) s += w[(long)b * 128 + t] * Hm[((long)b * 128 + t) * ND + d];
  outp[i] = s;
}

// sel_score[b,c] = sum_d tanh(P1+P2)*w[d] + bias ; c-mask -> NEG
__global__ void k_sel_score(const float* __restrict__ P1, const float* __restrict__ P2,
                            const float* __restrict__ w, const float* __restrict__ bw,
                            const int* __restrict__ clen, float* __restrict__ outp) {
  int i = blockIdx.x * blockDim.x + threadIdx.x;
  if (i >= NB * NTC) return;
  int b = i >> 5, c = i & 31;
  const float* p1 = P1 + (long)i * ND;
  const float* p2 = P2 + (long)i * ND;
  float s = 0.f;
  for (int d = 0; d < ND; ++d) s += tanhf(p1[d] + p2[d]) * w[d];
  s += bw[0];
  if (c >= clen[b]) s = NEGV;
  outp[i] = s;
}

// cond_num_score[b,j] = sum_d tanh(Q[b,d]) * Wn[j,d] + bn[j]   (j<5)
__global__ void k_num_score(const float* __restrict__ Q, const float* __restrict__ Wn,
                            const float* __restrict__ bn, float* __restrict__ outp) {
  int i = blockIdx.x * blockDim.x + threadIdx.x;
  if (i >= NB * 5) return;
  int b = i / 5, j = i % 5;
  const float* q = Q + (long)b * ND;
  const float* wj = Wn + (long)j * ND;
  float s = 0.f;
  for (int d = 0; d < ND; ++d) s += tanhf(q[d]) * wj[d];
  outp[i] = s + bn[j];
}

// cond_col_score[b,c] = sum_d relu(G1+G2+G3[b])*w[d] + bias ; c-mask -> NEG
__global__ void k_cc_score(const float* __restrict__ G1, const float* __restrict__ G2,
                           const float* __restrict__ G3, const float* __restrict__ w,
                           const float* __restrict__ bw, const int* __restrict__ clen,
                           float* __restrict__ outp) {
  int i = blockIdx.x * blockDim.x + threadIdx.x;
  if (i >= NB * NTC) return;
  int b = i >> 5, c = i & 31;
  const float* g1 = G1 + (long)i * ND;
  const float* g2 = G2 + (long)i * ND;
  const float* g3 = G3 + (long)b * ND;
  float s = 0.f;
  for (int d = 0; d < ND; ++d) s += fmaxf(g1[d] + g2[d] + g3[d], 0.f) * w[d];
  s += bw[0];
  if (c >= clen[b]) s = NEGV;
  outp[i] = s;
}

// ---------------------------------------------------------------------------
// Host orchestration
// ---------------------------------------------------------------------------
extern "C" void kernel_launch(void* const* d_in, const int* in_sizes, int n_in,
                              void* d_out, int out_size, void* d_ws, size_t ws_size,
                              hipStream_t stream) {
  (void)in_sizes; (void)n_in; (void)out_size; (void)ws_size;

  const float* x_emb   = (const float*)d_in[0];
  const float* x_type  = (const float*)d_in[1];
  const float* col_inp = (const float*)d_in[2];
  const int*   x_len   = (const int*)d_in[3];
  const int*   col_len = (const int*)d_in[4];
  const int*   gt_sel  = (const int*)d_in[5];
  float* outp = (float*)d_out;

  char* wsb = (char*)d_ws;
  size_t off = 0;
  auto alloc = [&](size_t bytes) -> char* {
    char* p = wsb + off;
    off += (bytes + 255) & ~(size_t)255;
    return p;
  };
  auto cvt = [&](const void* src, size_t n) -> _Float16* {
    _Float16* p = (_Float16*)alloc(n * 2);
    k_f32_to_f16<<<dim3((unsigned)((n + 255) / 256)), dim3(256), 0, stream>>>((const float*)src, p, (int)n);
    return p;
  };
  auto gemm = [&](const _Float16* A, const _Float16* Bm, const float* bias, float* C,
                  int M, int N, int K, int lda, int ldb, int ldc,
                  long sA, long sB, long sC, int nb, int b_kxn) {
    dim3 g((N + BN - 1) / BN, (M + BM - 1) / BM, nb);
    k_gemm<<<g, dim3(128), 0, stream>>>(A, Bm, bias, C, M, N, K, lda, ldb, ldc, sA, sB, sC, b_kxn);
  };

  // ---- activation scratch ----
  _Float16* xcat_h   = (_Float16*)alloc((size_t)NB * NTX * 1024 * 2);
  _Float16* seq_h    = (_Float16*)alloc((size_t)NB * NTX * ND * 2);
  _Float16* colx_h   = (_Float16*)alloc((size_t)NB * NTC * ND * 2);
  _Float16* colseq_h = (_Float16*)alloc((size_t)NB * NTC * ND * 2);
  float*    gates    = (float*)alloc((size_t)NB * NTX * NG * 4);
  float*    henc     = (float*)alloc((size_t)NB * NTX * ND * 4);   // layer out / h_enc f32
  float*    ecol     = (float*)alloc((size_t)NB * NTC * ND * 4);   // e_col f32
  _Float16* henc_h   = (_Float16*)alloc((size_t)NB * NTX * ND * 2);
  _Float16* ecol_h   = (_Float16*)alloc((size_t)NB * NTC * ND * 2);
  float*    proj     = (float*)alloc((size_t)NB * NTX * ND * 4);
  _Float16* proj_h   = (_Float16*)alloc((size_t)NB * NTX * ND * 2);
  float*    attv     = (float*)alloc((size_t)NB * NTC * 128 * 4);
  _Float16* att_h    = (_Float16*)alloc((size_t)NB * NTC * 128 * 2);
  float*    Kc       = (float*)alloc((size_t)NB * NTC * ND * 4);
  _Float16* Kc_h     = (_Float16*)alloc((size_t)NB * NTC * ND * 2);
  float*    P1       = (float*)alloc((size_t)NB * NTC * ND * 4);
  float*    P2       = (float*)alloc((size_t)NB * NTC * ND * 4);
  float*    svec     = (float*)alloc((size_t)NB * 128 * 4);
  float*    chosen   = (float*)alloc((size_t)NB * ND * 4);
  float*    Kagg     = (float*)alloc((size_t)NB * ND * 4);
  float*    Knum     = (float*)alloc((size_t)NB * ND * 4);
  float*    Qnum     = (float*)alloc((size_t)NB * ND * 4);
  float*    G3       = (float*)alloc((size_t)NB * ND * 4);

  // ---- weights -> f16 ----
  _Float16* Wi_sc0[2] = { cvt(d_in[6], 1024 * 1024), cvt(d_in[9], 1024 * 1024) };
  _Float16* Wh_sc0[2] = { cvt(d_in[7], 1024 * 256),  cvt(d_in[10], 1024 * 256) };
  const float* b_sc0[2] = { (const float*)d_in[8], (const float*)d_in[11] };
  _Float16* Wi_sc1[2] = { cvt(d_in[12], 1024 * 512), cvt(d_in[15], 1024 * 512) };
  _Float16* Wh_sc1[2] = { cvt(d_in[13], 1024 * 256), cvt(d_in[16], 1024 * 256) };
  const float* b_sc1[2] = { (const float*)d_in[14], (const float*)d_in[17] };
  _Float16* Wi_co0[2] = { cvt(d_in[18], 1024 * 512), cvt(d_in[21], 1024 * 512) };
  _Float16* Wh_co0[2] = { cvt(d_in[19], 1024 * 256), cvt(d_in[22], 1024 * 256) };
  const float* b_co0[2] = { (const float*)d_in[20], (const float*)d_in[23] };
  _Float16* Wi_co1[2] = { cvt(d_in[24], 1024 * 512), cvt(d_in[27], 1024 * 512) };
  _Float16* Wh_co1[2] = { cvt(d_in[25], 1024 * 256), cvt(d_in[28], 1024 * 256) };
  const float* b_co1[2] = { (const float*)d_in[26], (const float*)d_in[29] };

  _Float16* W_tynum  = cvt(d_in[30], 512 * 512); const float* b_tynum  = (const float*)d_in[31];
  _Float16* W_numatt = cvt(d_in[32], 512 * 512); const float* b_numatt = (const float*)d_in[33];
  _Float16* W_selatt = cvt(d_in[34], 512 * 512); const float* b_selatt = (const float*)d_in[35];
  _Float16* W_selK   = cvt(d_in[36], 512 * 512); const float* b_selK   = (const float*)d_in[37];
  _Float16* W_selC   = cvt(d_in[38], 512 * 512); const float* b_selC   = (const float*)d_in[39];
  _Float16* W_ccatt  = cvt(d_in[40], 512 * 512); const float* b_ccatt  = (const float*)d_in[41];
  _Float16* W_ccK    = cvt(d_in[42], 512 * 512); const float* b_ccK    = (const float*)d_in[43];
  _Float16* W_ccC    = cvt(d_in[44], 512 * 512); const float* b_ccC    = (const float*)d_in[45];
  _Float16* W_ccS    = cvt(d_in[46], 512 * 512); const float* b_ccS    = (const float*)d_in[47];
  _Float16* W_colatt = cvt(d_in[48], 512 * 512); const float* b_colatt = (const float*)d_in[49];
  const float* W_numout = (const float*)d_in[50]; const float* b_numout = (const float*)d_in[51];
  const float* W_selout = (const float*)d_in[52]; const float* b_selout = (const float*)d_in[53];
  const float* W_ccout  = (const float*)d_in[54]; const float* b_ccout  = (const float*)d_in[55];

  // ---- build x_cat (f16) and col input (f16) ----
  {
    int n = NB * NTX * 1024;
    k_concat_h<<<dim3((n + 255) / 256), dim3(256), 0, stream>>>(x_emb, x_type, xcat_h, NB * NTX, 512);
    int m = NB * NTC * ND;
    k_f32_to_f16<<<dim3((m + 255) / 256), dim3(256), 0, stream>>>(col_inp, colx_h, m);
  }

  // ---- sequence biLSTM (sc_layers) ----
  for (int dir = 0; dir < 2; ++dir) {           // layer 0 (in=1024)
    gemm(xcat_h, Wi_sc0[dir], b_sc0[dir], gates, NB * NTX, NG, 1024, 1024, 1024, NG, 0, 0, 0, 1, 0);
    k_lstm_scan<<<dim3(NB / 16), dim3(256), 0, stream>>>(gates, Wh_sc0[dir], x_len, henc, NTX, dir * NHID, dir);
  }
  { int n = NB * NTX * ND;
    k_f32_to_f16<<<dim3((n + 255) / 256), dim3(256), 0, stream>>>(henc, seq_h, n); }
  for (int dir = 0; dir < 2; ++dir) {           // layer 1 (in=512)
    gemm(seq_h, Wi_sc1[dir], b_sc1[dir], gates, NB * NTX, NG, 512, 512, 512, NG, 0, 0, 0, 1, 0);
    k_lstm_scan<<<dim3(NB / 16), dim3(256), 0, stream>>>(gates, Wh_sc1[dir], x_len, henc, NTX, dir * NHID, dir);
  }
  { int n = NB * NTX * ND;
    k_f32_to_f16<<<dim3((n + 255) / 256), dim3(256), 0, stream>>>(henc, henc_h, n); }

  // ---- column biLSTM (col_layers) ----
  for (int dir = 0; dir < 2; ++dir) {
    gemm(colx_h, Wi_co0[dir], b_co0[dir], gates, NB * NTC, NG, 512, 512, 512, NG, 0, 0, 0, 1, 0);
    k_lstm_scan<<<dim3(NB / 16), dim3(256), 0, stream>>>(gates, Wh_co0[dir], col_len, ecol, NTC, dir * NHID, dir);
  }
  { int n = NB * NTC * ND;
    k_f32_to_f16<<<dim3((n + 255) / 256), dim3(256), 0, stream>>>(ecol, colseq_h, n); }
  for (int dir = 0; dir < 2; ++dir) {
    gemm(colseq_h, Wi_co1[dir], b_co1[dir], gates, NB * NTC, NG, 512, 512, 512, NG, 0, 0, 0, 1, 0);
    k_lstm_scan<<<dim3(NB / 16), dim3(256), 0, stream>>>(gates, Wh_co1[dir], col_len, ecol, NTC, dir * NHID, dir);
  }
  { int n = NB * NTC * ND;
    k_f32_to_f16<<<dim3((n + 255) / 256), dim3(256), 0, stream>>>(ecol, ecol_h, n); }

  const long sEC = (long)NTC * ND;     // e_col batch stride
  const long sHP = (long)NTX * ND;     // hproj / h_enc batch stride
  const long sAT = (long)NTC * 128;    // att batch stride
  const int nproj = NB * NTX * ND;
  const int natt  = NB * NTC * 128;
  const int nKc   = NB * NTC * ND;

  // ================= SEL head =================
  gemm(henc_h, W_selatt, b_selatt, proj, NB * NTX, ND, ND, ND, ND, ND, 0, 0, 0, 1, 0);
  k_f32_to_f16<<<dim3((nproj + 255) / 256), dim3(256), 0, stream>>>(proj, proj_h, nproj);
  gemm(ecol_h, proj_h, nullptr, attv, NTC, NTX, ND, ND, ND, NTX, sEC, sHP, sAT, NB, 0);
  k_softmax128<<<dim3(NB * NTC), dim3(128), 0, stream>>>(attv, x_len, nullptr, NTC);
  k_f32_to_f16<<<dim3((natt + 255) / 256), dim3(256), 0, stream>>>(attv, att_h, natt);
  gemm(att_h, henc_h, nullptr, Kc, NTC, ND, NTX, NTX, ND, ND, sAT, sHP, sEC, NB, 1);
  k_f32_to_f16<<<dim3((nKc + 255) / 256), dim3(256), 0, stream>>>(Kc, Kc_h, nKc);
  gemm(Kc_h, W_selK, b_selK, P1, NB * NTC, ND, ND, ND, ND, ND, 0, 0, 0, 1, 0);
  gemm(ecol_h, W_selC, b_selC, P2, NB * NTC, ND, ND, ND, ND, ND, 0, 0, 0, 1, 0);
  k_sel_score<<<dim3((NB * NTC + 255) / 256), dim3(256), 0, stream>>>(
      P1, P2, W_selout, b_selout, col_len, outp + NB * 5);

  // ================= COND-NUM head =================
  gemm(henc_h, W_numatt, b_numatt, proj, NB * NTX, ND, ND, ND, ND, ND, 0, 0, 0, 1, 0);
  k_f32_to_f16<<<dim3((nproj + 255) / 256), dim3(256), 0, stream>>>(proj, proj_h, nproj);
  gemm(ecol_h, proj_h, nullptr, attv, NTC, NTX, ND, ND, ND, NTX, sEC, sHP, sAT, NB, 0);
  k_softmax128<<<dim3(NB * NTC), dim3(128), 0, stream>>>(attv, x_len, col_len, NTC);
  k_colsum<<<dim3((NB * 128 + 255) / 256), dim3(256), 0, stream>>>(attv, svec);
  k_vecmat<<<dim3((NB * ND + 255) / 256), dim3(256), 0, stream>>>(svec, henc, Knum);
  { _Float16* Knum_h = (_Float16*)att_h;   // reuse (att consumed)
    int n = NB * ND;
    k_f32_to_f16<<<dim3((n + 255) / 256), dim3(256), 0, stream>>>(Knum, Knum_h, n);
    gemm(Knum_h, W_tynum, b_tynum, Qnum, NB, ND, ND, ND, ND, ND, 0, 0, 0, 1, 0); }
  k_num_score<<<dim3((NB * 5 + 63) / 64), dim3(64), 0, stream>>>(Qnum, W_numout, b_numout, outp);

  // ================= COND-COL head =================
  k_chosen<<<dim3(NB), dim3(ND), 0, stream>>>(ecol, gt_sel, chosen);
  gemm(henc_h, W_colatt, b_colatt, proj, NB * NTX, ND, ND, ND, ND, ND, 0, 0, 0, 1, 0);
  k_dotvec<<<dim3((NB * NTX + 255) / 256), dim3(256), 0, stream>>>(proj, chosen, svec, NTX, ND);
  k_softmax128<<<dim3(NB), dim3(128), 0, stream>>>(svec, x_len, nullptr, 1);
  k_vecmat<<<dim3((NB * ND + 255) / 256), dim3(256), 0, stream>>>(svec, henc, Kagg);
  gemm(henc_h, W_ccatt, b_ccatt, proj, NB * NTX, ND, ND, ND, ND, ND, 0, 0, 0, 1, 0);
  k_f32_to_f16<<<dim3((nproj + 255) / 256), dim3(256), 0, stream>>>(proj, proj_h, nproj);
  gemm(ecol_h, proj_h, nullptr, attv, NTC, NTX, ND, ND, ND, NTX, sEC, sHP, sAT, NB, 0);
  k_softmax128<<<dim3(NB * NTC), dim3(128), 0, stream>>>(attv, x_len, nullptr, NTC);
  k_f32_to_f16<<<dim3((natt + 255) / 256), dim3(256), 0, stream>>>(attv, att_h, natt);
  gemm(att_h, henc_h, nullptr, Kc, NTC, ND, NTX, NTX, ND, ND, sAT, sHP, sEC, NB, 1);
  k_f32_to_f16<<<dim3((nKc + 255) / 256), dim3(256), 0, stream>>>(Kc, Kc_h, nKc);
  gemm(Kc_h, W_ccK, b_ccK, P1, NB * NTC, ND, ND, ND, ND, ND, 0, 0, 0, 1, 0);
  gemm(ecol_h, W_ccC, b_ccC, P2, NB * NTC, ND, ND, ND, ND, ND, 0, 0, 0, 1, 0);
  { _Float16* Kagg_h = (_Float16*)att_h;   // reuse again (att consumed)
    int n = NB * ND;
    k_f32_to_f16<<<dim3((n + 255) / 256), dim3(256), 0, stream>>>(Kagg, Kagg_h, n);
    gemm(Kagg_h, W_ccS, b_ccS, G3, NB, ND, ND, ND, ND, ND, 0, 0, 0, 1, 0); }
  k_cc_score<<<dim3((NB * NTC + 255) / 256), dim3(256), 0, stream>>>(
      P1, P2, G3, W_ccout, b_ccout, col_len, outp + NB * 5 + NB * NTC);
}
